// OrbitWarsGNNPolicy_66108136620618
// MI455X (gfx1250) — matrix operand order repaired and checked by reading, not compiled, not run on hardware
//
#include <hip/hip_runtime.h>
#include <hip/hip_bf16.h>

typedef _Float16 half_t;
typedef __attribute__((ext_vector_type(16))) _Float16 v16h;
typedef __attribute__((ext_vector_type(8)))  _Float16 v8h;
typedef __attribute__((ext_vector_type(8)))  float    v8f;

#define B_   4
#define N_   256
#define H_   128
#define E_   64
#define NEGV (-10000.0f)
#define DIAG 141.42135623730951f

// ---------------- WMMA fragment helpers (CDNA5 16x16x32 f16, wave32) ----------------
__device__ __forceinline__ v16h cat8(v8h lo, v8h hi) {
  return __builtin_shufflevector(lo, hi, 0,1,2,3,4,5,6,7,8,9,10,11,12,13,14,15);
}
// A: 16x32 f16, row-major source; lane row = lane&15; hi=lane>>4 selects K subsets
__device__ __forceinline__ v16h load_a_frag(const half_t* row, int k0, int hi) {
  v8h lo = *(const v8h*)(row + k0 + hi * 8);
  v8h hh = *(const v8h*)(row + k0 + 16 + hi * 8);
  return cat8(lo, hh);
}
// B: 32x16 f16; source is B^T row-major [n][K] so K is contiguous per lane (col = lane&15)
__device__ __forceinline__ v16h load_b_frag(const half_t* colrow, int k0, int hi) {
  const half_t* p = colrow + k0 + hi * 8;
  v8h lo = *(const v8h*)(p);
  v8h hh = *(const v8h*)(p + 16);
  return cat8(lo, hh);
}
__device__ __forceinline__ v8f wmma32(v16h a, v16h b, v8f c) {
  return __builtin_amdgcn_wmma_f32_16x16x32_f16(false, a, false, b, (short)0, c, false, false);
}

// ---------------- weight prep: f32 [R][C] -> f16 transposed [C][R] ----------------
__global__ void convT(const float* __restrict__ src, half_t* __restrict__ dst, int R, int C) {
  int idx = blockIdx.x * blockDim.x + threadIdx.x;
  if (idx >= R * C) return;
  int r = idx / C, c = idx % C;
  dst[c * R + r] = (half_t)src[idx];
}

// ---------------- generic WMMA GEMM: A(MxK f16) @ Bt(NxK f16) -> f32/f16 ----------------
__global__ void gemm16(const half_t* __restrict__ A, const half_t* __restrict__ Bt,
                       float* outF, half_t* outH, const float* bias,
                       int M, int Nn, int K, int relu) {
  int wave = threadIdx.x >> 5;
  int lane = threadIdx.x & 31;
  int rlo = lane & 15, hi = lane >> 4;
  int tile = blockIdx.x * (blockDim.x >> 5) + wave;
  int ntn = Nn >> 4;
  int tm = tile / ntn, tn = tile % ntn;
  if (tm * 16 >= M) return;
  const half_t* arow = A + (size_t)(tm * 16 + rlo) * K;
  const half_t* brow = Bt + (size_t)(tn * 16 + rlo) * K;
  float bv = bias ? bias[tn * 16 + rlo] : 0.0f;
  v8f c;
#pragma unroll
  for (int r = 0; r < 8; ++r) c[r] = bv;
  for (int k0 = 0; k0 < K; k0 += 32) {
    v16h a = load_a_frag(arow, k0, hi);
    v16h b = load_b_frag(brow, k0, hi);
    c = wmma32(a, b, c);
  }
#pragma unroll
  for (int r = 0; r < 8; ++r) {
    int row = tm * 16 + r + hi * 8;
    int col = tn * 16 + rlo;
    float v = c[r];
    if (relu) v = fmaxf(v, 0.0f);
    if (outF) outF[(size_t)row * Nn + col] = v;
    if (outH) outH[(size_t)row * Nn + col] = (half_t)v;
  }
}

// ---------------- node encoder layer 1 (K=10, VALU) ----------------
__global__ void node_l1(const float* __restrict__ nf, const float* __restrict__ w0,
                        const float* __restrict__ b0, half_t* __restrict__ act16) {
  int row = blockIdx.x;   // 0..1023
  int c = threadIdx.x;    // 0..127
  __shared__ float xr[10];
  if (c < 10) xr[c] = nf[row * 10 + c];
  __syncthreads();
  float acc = b0[c];
#pragma unroll
  for (int k = 0; k < 10; ++k) acc = fmaf(xr[k], w0[k * 128 + c], acc);
  act16[row * 128 + c] = (half_t)fmaxf(acc, 0.0f);
}

// ---------------- edge features + edge encoder (layer0 VALU, layer1 WMMA) ----------------
__global__ void edge_enc(const float* __restrict__ pos, const float* __restrict__ w0,
                         const float* __restrict__ b0, const half_t* __restrict__ w1t,
                         const float* __restrict__ b1, half_t* __restrict__ ee16) {
  int bi = blockIdx.x;
  int b = bi >> 8, i = bi & 255;
  int tid = threadIdx.x;
  __shared__ half_t t0[N_ * E_];          // 32KB relu(layer0) as f16
  __shared__ float sw0[6 * 64];
  __shared__ float sb0[64], sb1[64];
  for (int k = tid; k < 384; k += 256) sw0[k] = w0[k];
  if (tid < 64) { sb0[tid] = b0[tid]; sb1[tid] = b1[tid]; }

  float pix = pos[(b * N_ + i) * 2 + 0];
  float piy = pos[(b * N_ + i) * 2 + 1];
  int j = tid;
  float pjx = pos[(b * N_ + j) * 2 + 0];
  float pjy = pos[(b * N_ + j) * 2 + 1];
  float dx = pjx - pix, dy = pjy - piy;
  float d2 = dx * dx + dy * dy;
  float dist = fmaxf(sqrtf(d2 + 1e-12f), 1e-6f);
  float sx = (i == j) ? 1.0f : dx;
  float sy = (i == j) ? 0.0f : dy;
  float rr = sqrtf(sx * sx + sy * sy);
  float sn, cs;
  if (rr < 1e-20f) { sn = 0.0f; cs = 1.0f; } else { sn = sy / rr; cs = sx / rr; }
  float wx = 50.0f - pix, wy = 50.0f - piy;
  float t = (wx * dx + wy * dy) / fmaxf(d2, 1e-12f);
  t = fminf(fmaxf(t, 0.0f), 1.0f);
  float clx = pix + t * dx - 50.0f;
  float cly = piy + t * dy - 50.0f;
  float sd = sqrtf(clx * clx + cly * cly + 1e-12f);
  float inter = (sd < 7.0f) ? 1.0f : 0.0f;
  float clear = fminf(fmaxf((sd - 7.0f) / DIAG, 0.0f), 1.0f);
  float raw[6] = { dist / DIAG, dist / 6.0f, sn, cs, inter, clear };
  __syncthreads();
  for (int c = 0; c < 64; ++c) {
    float acc = sb0[c];
#pragma unroll
    for (int k = 0; k < 6; ++k) acc = fmaf(raw[k], sw0[k * 64 + c], acc);
    t0[j * 64 + c] = (half_t)fmaxf(acc, 0.0f);
  }
  __syncthreads();
  // 256x64 @ 64x64 via WMMA, 8 waves x 8 tiles
  int wave = tid >> 5, lane = tid & 31, rlo = lane & 15, hi = lane >> 4;
  half_t* eebase = ee16 + (size_t)(b * N_ + i) * N_ * E_;
  for (int t8 = 0; t8 < 8; ++t8) {
    int tileId = wave * 8 + t8;
    int tm = tileId >> 2, tn = tileId & 3;
    const half_t* arow = t0 + (tm * 16 + rlo) * 64;
    const half_t* brow = w1t + (tn * 16 + rlo) * 64;
    float bv = sb1[tn * 16 + rlo];
    v8f c;
#pragma unroll
    for (int r = 0; r < 8; ++r) c[r] = bv;
    c = wmma32(load_a_frag(arow, 0, hi), load_b_frag(brow, 0, hi), c);
    c = wmma32(load_a_frag(arow, 32, hi), load_b_frag(brow, 32, hi), c);
#pragma unroll
    for (int r = 0; r < 8; ++r) {
      int row = tm * 16 + r + hi * 8;
      int col = tn * 16 + rlo;
      eebase[(size_t)row * 64 + col] = (half_t)c[r];   // ee = relu(...)@w1 + b1 (no relu)
    }
  }
}

// ---------------- GAT attention + aggregate + residual layernorm ----------------
__global__ void gat_layer(const half_t* __restrict__ Wh16, const half_t* __restrict__ ee16,
                          float* __restrict__ h, half_t* __restrict__ h16,
                          const float* __restrict__ a_src, const float* __restrict__ a_tgt,
                          const float* __restrict__ a_edge, const float* __restrict__ ln_g,
                          const float* __restrict__ ln_b) {
  int bi = blockIdx.x;
  int b = bi >> 8, i = bi & 255;
  int tid = threadIdx.x;
  __shared__ half_t whl[64 * H_];   // 16KB chunk of Wh
  __shared__ float sc[4 * N_];      // scores -> exp values
  __shared__ float atg[H_], aed[E_ * 4];
  __shared__ float ssrc[4], mxs[4], invs[4];
  __shared__ float xbuf[H_];
  aed[tid] = a_edge[tid];
  if (tid < 128) atg[tid] = a_tgt[tid];
  if (tid < 4) {
    float acc = 0.0f;
    const half_t* wr = Wh16 + (size_t)(b * N_ + i) * H_ + tid * 32;
    for (int d = 0; d < 32; ++d) acc = fmaf((float)wr[d], a_src[tid * 32 + d], acc);
    ssrc[tid] = acc;
  }
  __syncthreads();
  { // per-thread j: scores for 4 heads
    int j = tid;
    const half_t* wr = Wh16 + (size_t)(b * N_ + j) * H_;
    float st[4] = {0, 0, 0, 0};
    for (int c = 0; c < 128; ++c) st[c >> 5] = fmaf((float)wr[c], atg[c], st[c >> 5]);
    const half_t* er = ee16 + ((size_t)(b * N_ + i) * N_ + j) * E_;
    float se[4] = {0, 0, 0, 0};
    for (int e = 0; e < 64; ++e) {
      float v = (float)er[e];
#pragma unroll
      for (int hh = 0; hh < 4; ++hh) se[hh] = fmaf(v, aed[e * 4 + hh], se[hh]);
    }
#pragma unroll
    for (int hh = 0; hh < 4; ++hh) {
      float s = ssrc[hh] + st[hh] + se[hh];
      s = (s > 0.0f) ? s : 0.2f * s;     // leaky_relu 0.2
      sc[hh * N_ + j] = s;
    }
  }
  __syncthreads();
  if (tid < 4) {
    float m = -3.4e38f;
    for (int j = 0; j < N_; ++j) m = fmaxf(m, sc[tid * N_ + j]);
    mxs[tid] = m;
  }
  __syncthreads();
  {
    int j = tid;
#pragma unroll
    for (int hh = 0; hh < 4; ++hh) sc[hh * N_ + j] = __expf(sc[hh * N_ + j] - mxs[hh]);
  }
  __syncthreads();
  if (tid < 4) {
    float s = 0.0f;
    for (int j = 0; j < N_; ++j) s += sc[tid * N_ + j];
    invs[tid] = 1.0f / s;
  }
  // aggregation over j in 4 chunks of 64 rows staged through LDS
  float outacc = 0.0f;
  int c = tid, hh = tid >> 5;
  for (int chunk = 0; chunk < 4; ++chunk) {
    __syncthreads();
    const v8h* srcv = (const v8h*)(Wh16 + (size_t)(b * N_ + chunk * 64) * H_);
    v8h* dstv = (v8h*)whl;
    for (int idx = tid; idx < (64 * H_) / 8; idx += 256) dstv[idx] = srcv[idx];
    __syncthreads();
    if (tid < 128) {
      const float* scrow = sc + hh * N_ + chunk * 64;
      for (int jj = 0; jj < 64; ++jj)
        outacc = fmaf(scrow[jj], (float)whl[jj * H_ + c], outacc);
    }
  }
  __syncthreads();
  if (tid < 128) xbuf[c] = outacc * invs[hh] + h[(size_t)(b * N_ + i) * H_ + c];
  __syncthreads();
  if (tid < 128) {
    float m = 0.0f;
    for (int k = 0; k < 128; ++k) m += xbuf[k];
    m *= (1.0f / 128.0f);
    float v = 0.0f;
    for (int k = 0; k < 128; ++k) { float d = xbuf[k] - m; v = fmaf(d, d, v); }
    v *= (1.0f / 128.0f);
    float y = (xbuf[c] - m) * rsqrtf(v + 1e-5f) * ln_g[c] + ln_b[c];
    h[(size_t)(b * N_ + i) * H_ + c] = y;
    h16[(size_t)(b * N_ + i) * H_ + c] = (half_t)y;
  }
}

// ---------------- pool + g + tgvec/fgvec + noop logit ----------------
__global__ void pool_small(const float* __restrict__ h, const float* __restrict__ gp_w,
                           const float* __restrict__ gp_b, const float* __restrict__ tg_w0,
                           const float* __restrict__ tg_b0, const float* __restrict__ fr_w0,
                           const float* __restrict__ fr_b0, const float* __restrict__ no_w0,
                           const float* __restrict__ no_b0, const float* __restrict__ no_w1,
                           const float* __restrict__ no_b1,
                           float* __restrict__ tgvec, float* __restrict__ fgvec,
                           float* __restrict__ out) {
  int b = blockIdx.x, c = threadIdx.x;
  __shared__ float sm[128], sg[128], tno[64];
  float hm = 0.0f;
  for (int n = 0; n < N_; ++n) hm += h[(size_t)(b * N_ + n) * H_ + c];
  sm[c] = hm * (1.0f / N_);
  __syncthreads();
  float ga = gp_b[c];
  for (int k = 0; k < 128; ++k) ga = fmaf(sm[k], gp_w[k * 128 + c], ga);
  sg[c] = fmaxf(ga, 0.0f);
  __syncthreads();
  float tv = tg_b0[c], fv = fr_b0[c];
  for (int k = 0; k < 128; ++k) {
    tv = fmaf(sg[k], tg_w0[(320 + k) * 128 + c], tv);   // rows 2H+E..
    fv = fmaf(sg[k], fr_w0[(256 + k) * 128 + c], fv);   // rows 2H..
  }
  tgvec[b * 128 + c] = tv;
  fgvec[b * 128 + c] = fv;
  if (c < 64) {
    float a = no_b0[c];
    for (int k = 0; k < 128; ++k) a = fmaf(sg[k], no_w0[k * 64 + c], a);
    tno[c] = fmaxf(a, 0.0f);
  }
  __syncthreads();
  if (c == 0) {
    float a = no_b1[0];
    for (int e = 0; e < 64; ++e) a = fmaf(tno[e], no_w1[e], a);
    out[b * 257 + 256] = a;
  }
}

// ---------------- source logits ----------------
__global__ void src_logits_k(const float* __restrict__ h, const float* __restrict__ src_w,
                             const float* __restrict__ src_b, const float* __restrict__ owned,
                             float* __restrict__ out) {
  int b = blockIdx.x, n = threadIdx.x;
  float s = src_b[0];
  const float* hr = h + (size_t)(b * N_ + n) * H_;
  for (int c = 0; c < 128; ++c) s = fmaf(hr[c], src_w[c], s);
  if (owned[b * N_ + n] == 0.0f) s = NEGV;
  out[b * 257 + n] = s;
}

// ---------------- fused target head: te (WMMA) + ts/tt/tg + relu + dot(tg_w1) ----------------
__global__ void tgt_head(const half_t* __restrict__ ee16, const half_t* __restrict__ teBt,
                         const float* __restrict__ ts, const float* __restrict__ tt,
                         const float* __restrict__ tgvec, const float* __restrict__ tg_w1,
                         const float* __restrict__ tg_b1, float* __restrict__ out) {
  int bi = blockIdx.x;
  int b = bi >> 8, i = bi & 255;
  int tid = threadIdx.x;
  int wave = tid >> 5, lane = tid & 31, rlo = lane & 15, hi = lane >> 4;
  __shared__ float baset[128], w1s[128];
  if (tid < 128) {
    baset[tid] = ts[(size_t)(b * N_ + i) * H_ + tid] + tgvec[b * 128 + tid];
    w1s[tid] = tg_w1[tid];
  }
  __syncthreads();
  float bb1 = tg_b1[0];
  const half_t* eeb = ee16 + (size_t)(b * N_ + i) * N_ * E_;
  float* orow = out + 1028 + (size_t)(b * N_ + i) * N_;
  for (int jt = wave; jt < 16; jt += 8) {
    int j0 = jt * 16;
    const half_t* arow = eeb + (size_t)(j0 + rlo) * E_;
    v16h a0 = load_a_frag(arow, 0, hi);
    v16h a1 = load_a_frag(arow, 32, hi);
    float s[8];
#pragma unroll
    for (int r = 0; r < 8; ++r) s[r] = 0.0f;
    for (int nt = 0; nt < 8; ++nt) {
      int col = nt * 16 + rlo;
      const half_t* brow = teBt + (size_t)col * E_;
      v8f c;
#pragma unroll
      for (int r = 0; r < 8; ++r) c[r] = 0.0f;
      c = wmma32(a0, load_b_frag(brow, 0, hi), c);
      c = wmma32(a1, load_b_frag(brow, 32, hi), c);
      float bt = baset[col], wv = w1s[col];
#pragma unroll
      for (int r = 0; r < 8; ++r) {
        int row = j0 + r + hi * 8;
        float pre = c[r] + bt + tt[(size_t)(b * N_ + row) * H_ + col];
        pre = fmaxf(pre, 0.0f);
        s[r] = fmaf(pre, wv, s[r]);
      }
    }
#pragma unroll
    for (int r = 0; r < 8; ++r) {           // reduce over the 16 columns (lanes rlo)
      s[r] += __shfl_xor(s[r], 1, 32);
      s[r] += __shfl_xor(s[r], 2, 32);
      s[r] += __shfl_xor(s[r], 4, 32);
      s[r] += __shfl_xor(s[r], 8, 32);
    }
    if (rlo == 0) {
#pragma unroll
      for (int r = 0; r < 8; ++r) {
        int row = j0 + r + hi * 8;
        float v = s[r] + bb1;
        if (row == i) v = NEGV;
        orow[row] = v;
      }
    }
  }
}

// ---------------- frac head (no WMMA needed: pre_f = fs_i + ft_j + fg) ----------------
__global__ void frac_head(const float* __restrict__ fs, const float* __restrict__ ft,
                          const float* __restrict__ fgvec, const float* __restrict__ fr_w1,
                          const float* __restrict__ fr_b1, float* __restrict__ out) {
  int bi = blockIdx.x;
  int b = bi >> 8, i = bi & 255;
  int tid = threadIdx.x;
  __shared__ float basef[128];
  __shared__ float w1f[512];
  if (tid < 128) basef[tid] = fs[(size_t)(b * N_ + i) * H_ + tid] + fgvec[b * 128 + tid];
  for (int k = tid; k < 512; k += 256) w1f[k] = fr_w1[k];
  __syncthreads();
  int j = tid;
  float a0 = fr_b1[0], a1 = fr_b1[1], a2 = fr_b1[2], a3 = fr_b1[3];
  const float* fr = ft + (size_t)(b * N_ + j) * H_;
  for (int c = 0; c < 128; ++c) {
    float p = fmaxf(basef[c] + fr[c], 0.0f);
    a0 = fmaf(p, w1f[c * 4 + 0], a0);
    a1 = fmaf(p, w1f[c * 4 + 1], a1);
    a2 = fmaf(p, w1f[c * 4 + 2], a2);
    a3 = fmaf(p, w1f[c * 4 + 3], a3);
  }
  float* o = out + 263172 + ((size_t)(b * N_ + i) * N_ + j) * 4;
  o[0] = a0; o[1] = a1; o[2] = a2; o[3] = a3;
}

// ==================================================================================
extern "C" void kernel_launch(void* const* d_in, const int* in_sizes, int n_in,
                              void* d_out, int out_size, void* d_ws, size_t ws_size,
                              hipStream_t stream) {
  (void)in_sizes; (void)n_in; (void)out_size; (void)ws_size;
  // pytree (sorted-key) flatten order of setup_inputs
  const float* nf    = (const float*)d_in[0];
  const float* pos   = (const float*)d_in[1];
  const float* owned = (const float*)d_in[2];
  const float* ee_b0 = (const float*)d_in[3];
  const float* ee_b1 = (const float*)d_in[4];
  const float* ee_w0 = (const float*)d_in[5];
  const float* ee_w1 = (const float*)d_in[6];
  const float* fr_b0 = (const float*)d_in[7];
  const float* fr_b1 = (const float*)d_in[8];
  const float* fr_w0 = (const float*)d_in[9];
  const float* fr_w1 = (const float*)d_in[10];
  const float* g1W   = (const float*)d_in[11];
  const float* g1ae  = (const float*)d_in[12];
  const float* g1as  = (const float*)d_in[13];
  const float* g1at  = (const float*)d_in[14];
  const float* g1lnb = (const float*)d_in[15];
  const float* g1lng = (const float*)d_in[16];
  const float* g2W   = (const float*)d_in[17];
  const float* g2ae  = (const float*)d_in[18];
  const float* g2as  = (const float*)d_in[19];
  const float* g2at  = (const float*)d_in[20];
  const float* g2lnb = (const float*)d_in[21];
  const float* g2lng = (const float*)d_in[22];
  const float* gp_b  = (const float*)d_in[23];
  const float* gp_w  = (const float*)d_in[24];
  const float* ne_b0 = (const float*)d_in[25];
  const float* ne_b1 = (const float*)d_in[26];
  const float* ne_w0 = (const float*)d_in[27];
  const float* ne_w1 = (const float*)d_in[28];
  const float* no_b0 = (const float*)d_in[29];
  const float* no_b1 = (const float*)d_in[30];
  const float* no_w0 = (const float*)d_in[31];
  const float* no_w1 = (const float*)d_in[32];
  const float* src_b = (const float*)d_in[33];
  const float* src_w = (const float*)d_in[34];
  const float* tg_b0 = (const float*)d_in[35];
  const float* tg_b1 = (const float*)d_in[36];
  const float* tg_w0 = (const float*)d_in[37];
  const float* tg_w1 = (const float*)d_in[38];
  float* out = (float*)d_out;

  char* base = (char*)d_ws;
  size_t off = 0;
  auto alloc = [&](size_t bytes) -> char* {
    char* p = base + off;
    off = (off + bytes + 255) & ~(size_t)255;
    return p;
  };
  float*  h      = (float*)alloc((size_t)B_ * N_ * H_ * 4);
  half_t* h16    = (half_t*)alloc((size_t)B_ * N_ * H_ * 2);
  half_t* act16  = (half_t*)alloc((size_t)B_ * N_ * H_ * 2);
  half_t* Wh16   = (half_t*)alloc((size_t)B_ * N_ * H_ * 2);
  half_t* ee16   = (half_t*)alloc((size_t)B_ * N_ * N_ * E_ * 2);   // 32MB, L2-resident
  float*  ts_    = (float*)alloc((size_t)B_ * N_ * H_ * 4);
  float*  tt_    = (float*)alloc((size_t)B_ * N_ * H_ * 4);
  float*  fs_    = (float*)alloc((size_t)B_ * N_ * H_ * 4);
  float*  ft_    = (float*)alloc((size_t)B_ * N_ * H_ * 4);
  float*  tgvec  = (float*)alloc((size_t)B_ * H_ * 4);
  float*  fgvec  = (float*)alloc((size_t)B_ * H_ * 4);
  half_t* ne_w1t = (half_t*)alloc(128 * 128 * 2);
  half_t* ee_w1t = (half_t*)alloc(64 * 64 * 2);
  half_t* g1Wt   = (half_t*)alloc(128 * 128 * 2);
  half_t* g2Wt   = (half_t*)alloc(128 * 128 * 2);
  half_t* tsBt   = (half_t*)alloc(128 * 128 * 2);
  half_t* ttBt   = (half_t*)alloc(128 * 128 * 2);
  half_t* teBt   = (half_t*)alloc(128 * 64 * 2);
  half_t* fsBt   = (half_t*)alloc(128 * 128 * 2);
  half_t* ftBt   = (half_t*)alloc(128 * 128 * 2);

  // weight prep (f32 -> f16, transposed for B-fragments)
  convT<<<64, 256, 0, stream>>>(ne_w1, ne_w1t, 128, 128);
  convT<<<16, 256, 0, stream>>>(ee_w1, ee_w1t, 64, 64);
  convT<<<64, 256, 0, stream>>>(g1W, g1Wt, 128, 128);
  convT<<<64, 256, 0, stream>>>(g2W, g2Wt, 128, 128);
  convT<<<64, 256, 0, stream>>>(tg_w0, tsBt, 128, 128);
  convT<<<64, 256, 0, stream>>>(tg_w0 + 128 * 128, ttBt, 128, 128);
  convT<<<32, 256, 0, stream>>>(tg_w0 + 256 * 128, teBt, 64, 128);
  convT<<<64, 256, 0, stream>>>(fr_w0, fsBt, 128, 128);
  convT<<<64, 256, 0, stream>>>(fr_w0 + 128 * 128, ftBt, 128, 128);

  // node encoder
  node_l1<<<B_ * N_, 128, 0, stream>>>(nf, ne_w0, ne_b0, act16);
  gemm16<<<128, 128, 0, stream>>>(act16, ne_w1t, h, h16, ne_b1, 1024, 128, 128, 1);

  // edge features + encoder -> ee16 (f16)
  edge_enc<<<B_ * N_, 256, 0, stream>>>(pos, ee_w0, ee_b0, ee_w1t, ee_b1, ee16);

  // GAT layer 1
  gemm16<<<128, 128, 0, stream>>>(h16, g1Wt, nullptr, Wh16, nullptr, 1024, 128, 128, 0);
  gat_layer<<<B_ * N_, 256, 0, stream>>>(Wh16, ee16, h, h16, g1as, g1at, g1ae, g1lng, g1lnb);
  // GAT layer 2
  gemm16<<<128, 128, 0, stream>>>(h16, g2Wt, nullptr, Wh16, nullptr, 1024, 128, 128, 0);
  gat_layer<<<B_ * N_, 256, 0, stream>>>(Wh16, ee16, h, h16, g2as, g2at, g2ae, g2lng, g2lnb);

  // global pool, per-batch vectors, noop logit, source logits
  pool_small<<<B_, 128, 0, stream>>>(h, gp_w, gp_b, tg_w0, tg_b0, fr_w0, fr_b0,
                                     no_w0, no_b0, no_w1, no_b1, tgvec, fgvec, out);
  src_logits_k<<<B_, 256, 0, stream>>>(h, src_w, src_b, owned, out);

  // ts/tt/fs/ft projections
  gemm16<<<128, 128, 0, stream>>>(h16, tsBt, ts_, nullptr, nullptr, 1024, 128, 128, 0);
  gemm16<<<128, 128, 0, stream>>>(h16, ttBt, tt_, nullptr, nullptr, 1024, 128, 128, 0);
  gemm16<<<128, 128, 0, stream>>>(h16, fsBt, fs_, nullptr, nullptr, 1024, 128, 128, 0);
  gemm16<<<128, 128, 0, stream>>>(h16, ftBt, ft_, nullptr, nullptr, 1024, 128, 128, 0);

  // fused heads
  tgt_head<<<B_ * N_, 256, 0, stream>>>(ee16, teBt, ts_, tt_, tgvec, tg_w1, tg_b1, out);
  frac_head<<<B_ * N_, 256, 0, stream>>>(fs_, ft_, fgvec, fr_w1, fr_b1, out);
}
